// GATBase_50611894616410
// MI455X (gfx1250) — compile-verified, matrix operand order checked by reading, not run on hardware
//
#include <hip/hip_runtime.h>
#include <math.h>

typedef _Float16 v16h __attribute__((ext_vector_type(16)));
typedef _Float16 v4h  __attribute__((ext_vector_type(4)));
typedef float    v8f  __attribute__((ext_vector_type(8)));
typedef float    v4f  __attribute__((ext_vector_type(4)));

#define HID 768
#define ALPHA_F 0.5f

// ---- WMMA fragment loader: 16-bit A/B layout per CDNA5 ISA 7.12.2 ----
// lane 0-15 : row/col = lane,    K = [0..7] in elems 0..7,  K=[16..23] in elems 8..15
// lane 16-31: row/col = lane-16, K = [8..15] in elems 0..7, K=[24..31] in elems 8..15
__device__ __forceinline__ v16h load_frag16(const _Float16* rowptr, int khalf) {
  v16h f;
#pragma unroll
  for (int j = 0; j < 4; ++j) {
    f[2 * j]         = rowptr[khalf + 2 * j];
    f[2 * j + 1]     = rowptr[khalf + 2 * j + 1];
    f[8 + 2 * j]     = rowptr[khalf + 16 + 2 * j];
    f[8 + 2 * j + 1] = rowptr[khalf + 16 + 2 * j + 1];
  }
  return f;
}

// EP: 0 = +bias ; 1 = (gemm+bias)*(1-a) + fuseT*a ; 2 = tanh(gemm+bias)
// gatherA: optional row indirection (A row = A[gatherA[r]]), for GAT gather-GEMM.
// Block tile 128x128, 8 wave32s, each wave 32x64 (2x4 WMMA accs).
// Double-buffered LDS; next tile loaded to regs while WMMAs run on current.
template <int EP>
__global__ __launch_bounds__(256) void wmma_gemm_kernel(
    const float* __restrict__ A, const int* __restrict__ gatherA,
    const float* __restrict__ B, const float* __restrict__ bias,
    const float* __restrict__ fuseT, float* __restrict__ C,
    int M, int N, int K, int lda) {
  __shared__ _Float16 sA[2][128][40];  // [buf][row][k], +8 halves pad
  __shared__ _Float16 sB[2][128][40];  // [buf][col][k] (transposed stage)

  const int tid = threadIdx.x;
  const int rowBase = blockIdx.y * 128;
  const int colBase = blockIdx.x * 128;
  const int lane = tid & 31;
  const int wave = tid >> 5;
  const int wm = (wave & 3) * 32;   // wave row offset in block tile
  const int wn = (wave >> 2) * 64;  // wave col offset in block tile
  const int lrow = lane & 15;
  const int khalf = (lane >> 4) << 3;
  const bool ldaVec = ((lda & 3) == 0);

  // ---- loop-invariant staging descriptors (gather hoisted out of K loop) ----
  // A tile: 128 rows x 32 k = 1024 float4 items; item id -> row = id>>3, k4 = (id&7)*4
  const float* aptrs[4]; int arow_l[4], akoff[4];
#pragma unroll
  for (int j = 0; j < 4; ++j) {
    int id = tid + 256 * j;
    arow_l[j] = id >> 3;
    akoff[j]  = (id & 7) * 4;
    int gr = rowBase + arow_l[j];
    if (gr < M) {
      long ar = gatherA ? (long)gatherA[gr] : (long)gr;
      aptrs[j] = A + ar * (long)lda;
    } else {
      aptrs[j] = nullptr;
    }
  }
  // B tile: 128 cols x 32 k = 1024 items; strided column reads (B is L2-resident)
  const float* bptrs[4]; int bcol_l[4], bkoff[4];
#pragma unroll
  for (int j = 0; j < 4; ++j) {
    int id = tid + 256 * j;
    bcol_l[j] = id >> 3;
    bkoff[j]  = (id & 7) * 4;
    int gc = colBase + bcol_l[j];
    bptrs[j] = (gc < N) ? (B + gc) : nullptr;
  }

  v4h rah[4], rbh[4];
  auto load_regs = [&](int kbase) {
#pragma unroll
    for (int j = 0; j < 4; ++j) {
      v4f v = {0.f, 0.f, 0.f, 0.f};
      if (aptrs[j]) {
        int kb = kbase + akoff[j];
        if (ldaVec && (kb + 4 <= K)) {
          v = *(const v4f*)(aptrs[j] + kb);          // global_load_b128
        } else {
#pragma unroll
          for (int e = 0; e < 4; ++e) if (kb + e < K) v[e] = aptrs[j][kb + e];
        }
      }
#pragma unroll
      for (int e = 0; e < 4; ++e) rah[j][e] = (_Float16)v[e];
    }
#pragma unroll
    for (int j = 0; j < 4; ++j) {
#pragma unroll
      for (int e = 0; e < 4; ++e) {
        int kk = kbase + bkoff[j] + e;
        float v = (bptrs[j] && kk < K) ? bptrs[j][(long)kk * N] : 0.f;
        rbh[j][e] = (_Float16)v;
      }
    }
  };
  auto store_lds = [&](int b) {
#pragma unroll
    for (int j = 0; j < 4; ++j) *(v4h*)(&sA[b][arow_l[j]][akoff[j]]) = rah[j];
#pragma unroll
    for (int j = 0; j < 4; ++j) *(v4h*)(&sB[b][bcol_l[j]][bkoff[j]]) = rbh[j];
  };

  v8f acc[2][4] = {};
  auto compute = [&](int b) {
    v16h a0 = load_frag16(&sA[b][wm + lrow][0], khalf);
    v16h a1 = load_frag16(&sA[b][wm + 16 + lrow][0], khalf);
#pragma unroll
    for (int ni = 0; ni < 4; ++ni) {
      v16h bf = load_frag16(&sB[b][wn + ni * 16 + lrow][0], khalf);
      acc[0][ni] = __builtin_amdgcn_wmma_f32_16x16x32_f16(false, a0, false, bf, (short)0, acc[0][ni], false, false);
      acc[1][ni] = __builtin_amdgcn_wmma_f32_16x16x32_f16(false, a1, false, bf, (short)0, acc[1][ni], false, false);
    }
  };

  // ---- software pipeline: regs(k+32) || wmma(k) with double-buffered LDS ----
  int buf = 0;
  load_regs(0);
  store_lds(0);
  __syncthreads();
  for (int k0 = 0; k0 < K; k0 += 32) {
    bool hasNext = (k0 + 32) < K;
    if (hasNext) load_regs(k0 + 32);   // global loads in flight during WMMAs
    compute(buf);
    if (hasNext) store_lds(buf ^ 1);
    __syncthreads();
    buf ^= 1;
  }

  // C/D layout: lane l -> col = l&15, rows r + 8*(l>>4), r=0..7
  const int rh = (lane >> 4) * 8;
#pragma unroll
  for (int mi = 0; mi < 2; ++mi)
#pragma unroll
    for (int ni = 0; ni < 4; ++ni)
#pragma unroll
      for (int r = 0; r < 8; ++r) {
        int gr = rowBase + wm + mi * 16 + rh + r;
        int gc = colBase + wn + ni * 16 + (lane & 15);
        if (gr < M && gc < N) {
          float v = acc[mi][ni][r];
          if (bias) v += bias[gc];
          if (EP == 1) v = v * (1.0f - ALPHA_F) + fuseT[(long)gr * N + gc] * ALPHA_F;
          if (EP == 2) v = tanhf(v);
          C[(long)gr * N + gc] = v;
        }
      }
}

// el[row] = h[row,:]@al ; er[row] = h[row,:]@ar
__global__ __launch_bounds__(256) void row_dots_kernel(
    const float* __restrict__ h, const float* __restrict__ al,
    const float* __restrict__ ar, float* __restrict__ el, float* __restrict__ er) {
  __shared__ float s1[256], s2[256];
  int row = blockIdx.x;
  const float* hp = h + (long)row * HID;
  float a = 0.f, b = 0.f;
  for (int c = threadIdx.x; c < HID; c += 256) {
    float v = hp[c];
    a += v * al[c];
    b += v * ar[c];
  }
  s1[threadIdx.x] = a; s2[threadIdx.x] = b;
  __syncthreads();
  for (int off = 128; off > 0; off >>= 1) {
    if (threadIdx.x < off) {
      s1[threadIdx.x] += s1[threadIdx.x + off];
      s2[threadIdx.x] += s2[threadIdx.x + off];
    }
    __syncthreads();
  }
  if (threadIdx.x == 0) { el[row] = s1[0]; er[row] = s2[0]; }
}

__global__ void fill_kernel(float* p, float v, long n) {
  long i = (long)blockIdx.x * blockDim.x + threadIdx.x;
  if (i < n) p[i] = v;
}

__global__ void fill4_kernel(v4f* p, float v, long n4) {
  long i = (long)blockIdx.x * blockDim.x + threadIdx.x;
  if (i < n4) p[i] = (v4f){v, v, v, v};
}

// ordered-int float atomic max (init target to -inf)
__device__ __forceinline__ void atomicMaxF(float* addr, float val) {
  if (val >= 0.f) atomicMax((int*)addr, __float_as_int(val));
  else            atomicMin((unsigned int*)addr, __float_as_uint(val));
}

__global__ void edge_max_kernel(const float* __restrict__ el, const float* __restrict__ er,
                                const int* __restrict__ src, const int* __restrict__ dst,
                                float* __restrict__ eBuf, float* __restrict__ m, int E) {
  int i = blockIdx.x * blockDim.x + threadIdx.x;
  if (i >= E) return;
  float e = el[src[i]] + er[dst[i]];
  e = (e >= 0.f) ? e : 0.2f * e;  // leaky_relu, NEG_SLOPE=0.2
  eBuf[i] = e;
  atomicMaxF(&m[dst[i]], e);
}

__global__ void edge_exp_kernel(const float* __restrict__ m, const int* __restrict__ dst,
                                float* __restrict__ eBuf, float* __restrict__ s, int E) {
  int i = blockIdx.x * blockDim.x + threadIdx.x;
  if (i >= E) return;
  int d = dst[i];
  float a = expf(eBuf[i] - m[d]);
  eBuf[i] = a;
  atomicAdd(&s[d], a);
}

// one block per edge; accumulate alpha * h[src] into out[dst]
__global__ __launch_bounds__(256) void edge_accum_kernel(
    const float* __restrict__ h, const float* __restrict__ aBuf,
    const float* __restrict__ s, const int* __restrict__ src,
    const int* __restrict__ dst, float* __restrict__ out, int E) {
  int e = blockIdx.x;
  if (e >= E) return;
  int sN = src[e], dN = dst[e];
  float alpha = aBuf[e] / s[dN];
  const float* hp = h + (long)sN * HID;
  float* op = out + (long)dN * HID;
  for (int c = threadIdx.x; c < HID; c += blockDim.x)
    atomicAdd(&op[c], alpha * hp[c]);
}

// node[idx[row]] = gout[row], 16B at a time
__global__ void scatter4_kernel(const int* __restrict__ idx, const v4f* __restrict__ gout,
                                float* __restrict__ node, long total4) {
  long i = (long)blockIdx.x * blockDim.x + threadIdx.x;
  if (i >= total4) return;
  long row = i / (HID / 4);
  int c4 = (int)(i - row * (HID / 4));
  ((v4f*)node)[(long)idx[row] * (HID / 4) + c4] = gout[i];
}

extern "C" void kernel_launch(void* const* d_in, const int* in_sizes, int n_in,
                              void* d_out, int out_size, void* d_ws, size_t ws_size,
                              hipStream_t stream) {
  (void)in_sizes; (void)n_in; (void)out_size; (void)ws_size;
  const int U = 6000, O = 15000, I = 30000, SC = 2000, Cc = 15000;
  const int NN = U + O + I + SC + Cc;       // 68000
  const int IMG = 2048, UHID = 10, HASH = 256;

  const float* item_table    = (const float*)d_in[0];
  const float* item_proj_w   = (const float*)d_in[1];
  const float* item_proj_b   = (const float*)d_in[2];
  const float* item_graph    = (const float*)d_in[3];
  const float* outfit_table  = (const float*)d_in[4];
  const float* outfit_proj_w = (const float*)d_in[5];
  const float* outfit_proj_b = (const float*)d_in[6];
  const float* outfit_graph  = (const float*)d_in[7];
  const float* user_table    = (const float*)d_in[8];
  const float* user_proj_w   = (const float*)d_in[9];
  const float* user_proj_b   = (const float*)d_in[10];
  const float* user_graph    = (const float*)d_in[11];
  const float* scene_table   = (const float*)d_in[12];
  const float* scene_proj_w  = (const float*)d_in[13];
  const float* scene_proj_b  = (const float*)d_in[14];
  const float* scene_graph   = (const float*)d_in[15];
  const float* comb_table    = (const float*)d_in[16];
  const float* comb_proj_w   = (const float*)d_in[17];
  const float* comb_proj_b   = (const float*)d_in[18];
  const float* gat_w         = (const float*)d_in[19];
  const float* gat_al        = (const float*)d_in[20];
  const float* gat_ar        = (const float*)d_in[21];
  const float* hash_w        = (const float*)d_in[22];
  const float* hash_b        = (const float*)d_in[23];
  // d_in[24..28] are identity aranges (user/outfit/item/scene/comb idx) -> no-op gathers
  const int* co_index = (const int*)d_in[29];
  const int* oi_index = (const int*)d_in[30];
  const int* uc_index = (const int*)d_in[31];
  const int* sc_index = (const int*)d_in[32];
  const int* co_src = (const int*)d_in[33];
  const int* co_dst = (const int*)d_in[34];
  const int* oi_src = (const int*)d_in[35];
  const int* oi_dst = (const int*)d_in[36];
  const int* uc_src = (const int*)d_in[37];
  const int* uc_dst = (const int*)d_in[38];
  const int* sc_src = (const int*)d_in[39];
  const int* sc_dst = (const int*)d_in[40];

  // workspace carve-up (floats)
  const int MAXN = O + I;  // 45000 (largest subset); E_max = 5*MAXN
  float* ws   = (float*)d_ws;
  float* node = ws;  ws += (long)NN * HID;      // 68000x768
  float* hbuf = ws;  ws += (long)MAXN * HID;    // GAT h
  float* gout = ws;  ws += (long)MAXN * HID;    // GAT aggregation
  float* el = ws;    ws += MAXN;
  float* er = ws;    ws += MAXN;
  float* mbuf = ws;  ws += MAXN;
  float* sbuf = ws;  ws += MAXN;
  float* ebuf = ws;                              // 5*MAXN edge scratch

  dim3 blk(256);
  auto grid_for = [](int M, int N) { return dim3((unsigned)((N + 127) / 128), (unsigned)((M + 127) / 128)); };

  // node order: user | outfit | item | scene | comb
  float* nodeU = node;
  float* nodeO = node + (long)U * HID;
  float* nodeI = node + (long)(U + O) * HID;
  float* nodeS = node + (long)(U + O + I) * HID;
  float* nodeC = node + (long)(U + O + I + SC) * HID;

  // projections (EP=1 fuses alpha-blend with graph tables; comb EP=0)
  wmma_gemm_kernel<1><<<grid_for(U, HID), blk, 0, stream>>>(user_table,   nullptr, user_proj_w,   user_proj_b,   user_graph,   nodeU, U,  HID, UHID, UHID);
  wmma_gemm_kernel<1><<<grid_for(O, HID), blk, 0, stream>>>(outfit_table, nullptr, outfit_proj_w, outfit_proj_b, outfit_graph, nodeO, O,  HID, IMG,  IMG);
  wmma_gemm_kernel<1><<<grid_for(I, HID), blk, 0, stream>>>(item_table,   nullptr, item_proj_w,   item_proj_b,   item_graph,   nodeI, I,  HID, IMG,  IMG);
  wmma_gemm_kernel<1><<<grid_for(SC, HID), blk, 0, stream>>>(scene_table, nullptr, scene_proj_w,  scene_proj_b,  scene_graph,  nodeS, SC, HID, IMG,  IMG);
  wmma_gemm_kernel<0><<<grid_for(Cc, HID), blk, 0, stream>>>(comb_table,  nullptr, comb_proj_w,   comb_proj_b,   nullptr,      nodeC, Cc, HID, HID,  HID);

  const int* tri_index[4] = {co_index, oi_index, uc_index, sc_index};
  const int* tri_src[4]   = {co_src, oi_src, uc_src, sc_src};
  const int* tri_dst[4]   = {co_dst, oi_dst, uc_dst, sc_dst};
  const int  tri_n[4]     = {Cc + O, O + I, U + Cc, SC + Cc};
  const int  order[5]     = {1, 0, 3, 2, 1};   // ITER_IDS

  for (int it = 0; it < 5; ++it) {
    int g = order[it];
    int n = tri_n[g];
    int E = n * 5;            // DEG=4 random edges + self loops
    long feat4 = (long)n * (HID / 4);

    // h = node[idx] @ gat_w  (gather fused into GEMM A-load, hoisted out of K loop)
    wmma_gemm_kernel<0><<<grid_for(n, HID), blk, 0, stream>>>(node, tri_index[g], gat_w, nullptr, nullptr, hbuf, n, HID, HID, HID);
    row_dots_kernel<<<n, blk, 0, stream>>>(hbuf, gat_al, gat_ar, el, er);

    fill_kernel<<<(n + 255) / 256, blk, 0, stream>>>(mbuf, -INFINITY, n);
    fill_kernel<<<(n + 255) / 256, blk, 0, stream>>>(sbuf, 0.f, n);
    fill4_kernel<<<(unsigned)((feat4 + 255) / 256), blk, 0, stream>>>((v4f*)gout, 0.f, feat4);

    edge_max_kernel<<<(E + 255) / 256, blk, 0, stream>>>(el, er, tri_src[g], tri_dst[g], ebuf, mbuf, E);
    edge_exp_kernel<<<(E + 255) / 256, blk, 0, stream>>>(mbuf, tri_dst[g], ebuf, sbuf, E);
    edge_accum_kernel<<<E, blk, 0, stream>>>(hbuf, ebuf, sbuf, tri_src[g], tri_dst[g], gout, E);
    scatter4_kernel<<<(unsigned)((feat4 + 255) / 256), blk, 0, stream>>>(tri_index[g], (const v4f*)gout, node, feat4);
  }

  // hash head: tanh(node @ hash_w + hash_b) -> d_out [68000 x 256] f32
  wmma_gemm_kernel<2><<<grid_for(NN, HASH), blk, 0, stream>>>(node, nullptr, hash_w, hash_b, nullptr, (float*)d_out, NN, HASH, HID, HID);
}
